// GPT2Model_8924942041686
// MI455X (gfx1250) — matrix operand order repaired and checked by reading, not compiled, not run on hardware
//
#include <hip/hip_runtime.h>
#include <cstddef>
#include <cstdint>

// ---------------------------------------------------------------------------
// GPT-2 forward (L=6, B=2, T=1024, D=1024, H=16, DH=64, F=4096) for gfx1250.
// - all matmul FLOPs on v_wmma_f32_16x16x32_f16 (f32 accumulate)
// - weights pre-transposed to f16 [N][K] so BOTH gemm tiles are row-contiguous
//   -> staged with GLOBAL_LOAD_ASYNC_TO_LDS_B128 (ASYNCcnt), double-buffered
// - fused flash-attention (no materialized TxT scores)
// - bias/GELU/residual fused into compile-time-specialized GEMM epilogues
// ---------------------------------------------------------------------------

#define L_LAYERS 6
#define SEQ_T    1024
#define BATCH_B  2
#define DIM_D    1024
#define HEADS_H  16
#define DH       64
#define FFN_F    4096
#define BT       (BATCH_B * SEQ_T)   // 2048 rows
#define LN_EPS   1e-5f

typedef __attribute__((ext_vector_type(16))) _Float16 v16h;
typedef __attribute__((ext_vector_type(8)))  _Float16 v8h;
typedef __attribute__((ext_vector_type(8)))  float    v8f;

union F16x16 { v16h v; v8h h[2]; };

__device__ __forceinline__ v8f wmma_f16(v16h a, v16h b, v8f c) {
    return __builtin_amdgcn_wmma_f32_16x16x32_f16(
        /*neg_a=*/false, a, /*neg_b=*/false, b,
        /*c_mod=*/(short)0, c, /*reuse_a=*/false, /*reuse_b=*/false);
}

__device__ __forceinline__ float gelu_exact(float x) {
    return 0.5f * x * (1.0f + erff(x * 0.70710678118654752f));
}

// Async 16B global -> LDS copy (per lane), tracked by ASYNCcnt.
__device__ __forceinline__ void async_copy_b128(const _Float16* g, _Float16* l) {
    const unsigned lds = (unsigned)(size_t)l;   // low 32 bits = LDS offset
    asm volatile("global_load_async_to_lds_b128 %0, %1, off"
                 :: "v"(lds), "v"(g) : "memory");
}
__device__ __forceinline__ void wait_async0() {
    asm volatile("s_wait_asynccnt 0x0" ::: "memory");
}

// ---------------------------------------------------------------------------
// Embedding: x[b,t,:] = wte[id] + wpe[t]   (fp32)
// ---------------------------------------------------------------------------
__global__ void embed_kernel(const int* __restrict__ ids,
                             const float* __restrict__ wte,
                             const float* __restrict__ wpe,
                             float* __restrict__ x) {
    const int bt  = blockIdx.x;
    const int t   = bt & (SEQ_T - 1);
    const int id  = ids[bt];
    const int tid = threadIdx.x;
    const float4 a = ((const float4*)(wte + (size_t)id * DIM_D))[tid];
    const float4 p = ((const float4*)(wpe + (size_t)t  * DIM_D))[tid];
    float4 r; r.x = a.x + p.x; r.y = a.y + p.y; r.z = a.z + p.z; r.w = a.w + p.w;
    ((float4*)(x + (size_t)bt * DIM_D))[tid] = r;
}

// ---------------------------------------------------------------------------
// Weight convert + transpose: W[K][N] fp32 -> Wt[N][K] f16.
// 64x64 tiles through LDS; coalesced fp32 reads, coalesced f16 writes.
// grid = (N/64, K/64), 256 threads.
// ---------------------------------------------------------------------------
__global__ __launch_bounds__(256)
void cvt_t_kernel(const float* __restrict__ s, _Float16* __restrict__ d,
                  int K, int N) {
    __shared__ _Float16 tile[64][72];
    const int k0 = blockIdx.y * 64;
    const int n0 = blockIdx.x * 64;
    const int tr = threadIdx.x >> 4;          // 0..15
    const int tc = (threadIdx.x & 15) * 4;    // 0..60
    #pragma unroll
    for (int rr = 0; rr < 64; rr += 16) {
        const float4 v = *(const float4*)(s + (size_t)(k0 + tr + rr) * N + n0 + tc);
        tile[tr + rr][tc + 0] = (_Float16)v.x;
        tile[tr + rr][tc + 1] = (_Float16)v.y;
        tile[tr + rr][tc + 2] = (_Float16)v.z;
        tile[tr + rr][tc + 3] = (_Float16)v.w;
    }
    __syncthreads();
    #pragma unroll
    for (int rr = 0; rr < 64; rr += 16) {
        const int n = tr + rr;
        _Float16* o = d + (size_t)(n0 + n) * K + k0 + tc;
        o[0] = tile[tc + 0][n];
        o[1] = tile[tc + 1][n];
        o[2] = tile[tc + 2][n];
        o[3] = tile[tc + 3][n];
    }
}

// ---------------------------------------------------------------------------
// LayerNorm over D=1024. One block (256 thr) per row. Optional f16 / f32 out.
// ---------------------------------------------------------------------------
__global__ void ln_kernel(const float* __restrict__ x,
                          const float* __restrict__ g,
                          const float* __restrict__ b,
                          _Float16* __restrict__ out16,
                          float* __restrict__ out32) {
    const int row = blockIdx.x;
    const int tid = threadIdx.x;
    const float4 xv = ((const float4*)(x + (size_t)row * DIM_D))[tid];
    float s1 = xv.x + xv.y + xv.z + xv.w;
    float s2 = xv.x * xv.x + xv.y * xv.y + xv.z * xv.z + xv.w * xv.w;
    __shared__ float red[2][8];
    #pragma unroll
    for (int m = 16; m >= 1; m >>= 1) {
        s1 += __shfl_xor(s1, m, 32);
        s2 += __shfl_xor(s2, m, 32);
    }
    if ((tid & 31) == 0) { red[0][tid >> 5] = s1; red[1][tid >> 5] = s2; }
    __syncthreads();
    s1 = 0.f; s2 = 0.f;
    #pragma unroll
    for (int i = 0; i < 8; ++i) { s1 += red[0][i]; s2 += red[1][i]; }
    const float mean = s1 * (1.0f / DIM_D);
    const float inv  = rsqrtf(s2 * (1.0f / DIM_D) - mean * mean + LN_EPS);
    const float4 gv = ((const float4*)g)[tid];
    const float4 bv = ((const float4*)b)[tid];
    float4 y;
    y.x = (xv.x - mean) * inv * gv.x + bv.x;
    y.y = (xv.y - mean) * inv * gv.y + bv.y;
    y.z = (xv.z - mean) * inv * gv.z + bv.z;
    y.w = (xv.w - mean) * inv * gv.w + bv.w;
    const size_t base = (size_t)row * DIM_D + tid * 4;
    if (out32) {
        out32[base + 0] = y.x; out32[base + 1] = y.y;
        out32[base + 2] = y.z; out32[base + 3] = y.w;
    }
    if (out16) {
        out16[base + 0] = (_Float16)y.x; out16[base + 1] = (_Float16)y.y;
        out16[base + 2] = (_Float16)y.z; out16[base + 3] = (_Float16)y.w;
    }
}

// ---------------------------------------------------------------------------
// WMMA GEMM:  out = ACT(A[f16,MxK] * Bt[f16,NxK]^T + bias) (+ residual)
// A row-major [M][K]; Bt is the PRE-TRANSPOSED weight [N][K], so both LDS
// tiles ([m][k], [n][k]) are filled by contiguous async b128 copies.
// 128x128 tile, BK=32, 8 wave32 (2x4), 4x2 WMMA accs per wave.
// Double-buffered LDS: async fill of buf^1 overlaps WMMA on buf.
// ---------------------------------------------------------------------------
#define GP 40   // padded LDS row stride (f16): 80B = 5*16B
template <int ACT, bool RES, bool O32, bool O16>
__global__ __launch_bounds__(256)
void gemm_kernel(const _Float16* __restrict__ A,
                 const _Float16* __restrict__ Bt,
                 const float* __restrict__ bias,
                 const float* __restrict__ residual,
                 float* __restrict__ out32,
                 _Float16* __restrict__ out16,
                 int M, int N, int K) {
    __shared__ _Float16 As[2][128][GP];   // [buf][m][k]
    __shared__ _Float16 Bs[2][128][GP];   // [buf][n][k]

    const int m0   = blockIdx.y * 128;
    const int n0   = blockIdx.x * 128;
    const int lane = threadIdx.x & 31;
    const int wave = threadIdx.x >> 5;
    const int wm   = wave >> 2;          // 0..1
    const int wn   = wave & 3;           // 0..3
    const int lh   = lane >> 4;          // 0/1
    const int l15  = lane & 15;

    v8f acc[4][2] = {};

    // each thread copies 2 x 16B chunks for A and for B per K-step
    const int c0   = threadIdx.x * 2;
    const int row0 = c0 >> 2,       ko0 = (c0 & 3) * 8;
    const int row1 = (c0 + 1) >> 2, ko1 = ((c0 + 1) & 3) * 8;

    auto stage = [&](int k0, int buf) {
        async_copy_b128(A  + (size_t)(m0 + row0) * K + k0 + ko0, &As[buf][row0][ko0]);
        async_copy_b128(A  + (size_t)(m0 + row1) * K + k0 + ko1, &As[buf][row1][ko1]);
        async_copy_b128(Bt + (size_t)(n0 + row0) * K + k0 + ko0, &Bs[buf][row0][ko0]);
        async_copy_b128(Bt + (size_t)(n0 + row1) * K + k0 + ko1, &Bs[buf][row1][ko1]);
    };

    const int nk = K >> 5;
    stage(0, 0);
    wait_async0();
    __syncthreads();

    for (int kt = 0; kt < nk; ++kt) {
        const int cur = kt & 1;
        if (kt + 1 < nk) stage((kt + 1) << 5, cur ^ 1);

        F16x16 af[4];
        #pragma unroll
        for (int mt = 0; mt < 4; ++mt) {
            const int r = wm * 64 + mt * 16 + l15;
            af[mt].h[0] = *(const v8h*)&As[cur][r][lh * 8];
            af[mt].h[1] = *(const v8h*)&As[cur][r][16 + lh * 8];
        }
        F16x16 bf[2];
        #pragma unroll
        for (int nt = 0; nt < 2; ++nt) {
            const int c = wn * 32 + nt * 16 + l15;
            bf[nt].h[0] = *(const v8h*)&Bs[cur][c][lh * 8];
            bf[nt].h[1] = *(const v8h*)&Bs[cur][c][16 + lh * 8];
        }
        #pragma unroll
        for (int mt = 0; mt < 4; ++mt)
            #pragma unroll
            for (int nt = 0; nt < 2; ++nt)
                acc[mt][nt] = wmma_f16(af[mt].v, bf[nt].v, acc[mt][nt]);

        wait_async0();
        __syncthreads();
    }

    // ---- straight-line epilogue (compile-time specialized) ----
    #pragma unroll
    for (int nt = 0; nt < 2; ++nt) {
        const int col = n0 + wn * 32 + nt * 16 + l15;
        const float bv = bias[col];
        #pragma unroll
        for (int mt = 0; mt < 4; ++mt) {
            const int rowb = m0 + wm * 64 + mt * 16 + lh * 8;
            v8f a = acc[mt][nt];
            #pragma unroll
            for (int r = 0; r < 8; ++r) {
                float v = a[r] + bv;
                if (ACT == 1) v = gelu_exact(v);
                const size_t idx = (size_t)(rowb + r) * N + col;
                if (RES) v += residual[idx];
                if (O32) out32[idx] = v;
                if (O16) out16[idx] = (_Float16)v;
            }
        }
    }
}

// ---------------------------------------------------------------------------
// Fused attention (flash-style, no causal mask per reference).
// grid = (T/64, H, B), 128 threads = 4 waves; wave w owns 16 Q rows.
// q/k/v layout [B,T,H,DH] f16: QK^T fragments read straight from global;
// V LDS-transposed for P*V. Scale 1/8 folded into Q fragments.
// ---------------------------------------------------------------------------
#define AP 72   // 64 + 8 pad
__global__ __launch_bounds__(128)
void attn_kernel(const _Float16* __restrict__ q,
                 const _Float16* __restrict__ k,
                 const _Float16* __restrict__ v,
                 _Float16* __restrict__ o) {
    const int qt   = blockIdx.x;
    const int h    = blockIdx.y;
    const int b    = blockIdx.z;
    const int lane = threadIdx.x & 31;
    const int wave = threadIdx.x >> 5;
    const int lh   = lane >> 4;
    const int l15  = lane & 15;

    __shared__ _Float16 Vt[DH][AP];
    __shared__ _Float16 Ps[4][16][AP];

    const int qrow = qt * 64 + wave * 16 + l15;
    const _Float16* qb = q + (((size_t)b * SEQ_T + qrow) * HEADS_H + h) * DH;
    F16x16 aq[2];
    #pragma unroll
    for (int s = 0; s < 2; ++s) {
        aq[s].h[0] = *(const v8h*)(qb + s * 32 + lh * 8);
        aq[s].h[1] = *(const v8h*)(qb + s * 32 + 16 + lh * 8);
        #pragma unroll
        for (int e = 0; e < 16; ++e) aq[s].v[e] = aq[s].v[e] * (_Float16)0.125f;
    }

    float m_i[8], l_i[8];
    v8f   o_acc[4] = {};
    #pragma unroll
    for (int r = 0; r < 8; ++r) { m_i[r] = -1e30f; l_i[r] = 0.0f; }

    for (int ktile = 0; ktile < SEQ_T / 64; ++ktile) {
        const int kt0 = ktile * 64;
        __syncthreads();
        {
            const int key = threadIdx.x & 63;
            const int dhh = (threadIdx.x >> 6) & 1;
            const _Float16* vb =
                v + (((size_t)b * SEQ_T + kt0 + key) * HEADS_H + h) * DH + dhh * 32;
            v8h v0 = ((const v8h*)vb)[0];
            v8h v1 = ((const v8h*)vb)[1];
            v8h v2 = ((const v8h*)vb)[2];
            v8h v3 = ((const v8h*)vb)[3];
            #pragma unroll
            for (int j = 0; j < 8; ++j) Vt[dhh * 32 + j][key]      = v0[j];
            #pragma unroll
            for (int j = 0; j < 8; ++j) Vt[dhh * 32 + 8 + j][key]  = v1[j];
            #pragma unroll
            for (int j = 0; j < 8; ++j) Vt[dhh * 32 + 16 + j][key] = v2[j];
            #pragma unroll
            for (int j = 0; j < 8; ++j) Vt[dhh * 32 + 24 + j][key] = v3[j];
        }
        __syncthreads();

        v8f s_acc[4] = {};
        #pragma unroll
        for (int s = 0; s < 2; ++s) {
            #pragma unroll
            for (int nt = 0; nt < 4; ++nt) {
                const int key = kt0 + nt * 16 + l15;
                const _Float16* kb =
                    k + (((size_t)b * SEQ_T + key) * HEADS_H + h) * DH + s * 32 + lh * 8;
                F16x16 bk;
                bk.h[0] = ((const v8h*)kb)[0];
                bk.h[1] = ((const v8h*)kb)[2];
                s_acc[nt] = wmma_f16(aq[s].v, bk.v, s_acc[nt]);
            }
        }

        float alpha[8];
        #pragma unroll
        for (int r = 0; r < 8; ++r) {
            float mx = s_acc[0][r];
            mx = fmaxf(mx, s_acc[1][r]);
            mx = fmaxf(mx, s_acc[2][r]);
            mx = fmaxf(mx, s_acc[3][r]);
            #pragma unroll
            for (int msk = 8; msk >= 1; msk >>= 1)
                mx = fmaxf(mx, __shfl_xor(mx, msk, 32));
            const float mnew = fmaxf(m_i[r], mx);
            alpha[r] = __expf(m_i[r] - mnew);
            float rs = 0.0f;
            #pragma unroll
            for (int nt = 0; nt < 4; ++nt) {
                const float p = __expf(s_acc[nt][r] - mnew);
                s_acc[nt][r] = p;
                rs += p;
            }
            #pragma unroll
            for (int msk = 8; msk >= 1; msk >>= 1)
                rs += __shfl_xor(rs, msk, 32);
            l_i[r] = l_i[r] * alpha[r] + rs;
            m_i[r] = mnew;
        }
        #pragma unroll
        for (int nt = 0; nt < 4; ++nt)
            #pragma unroll
            for (int r = 0; r < 8; ++r)
                o_acc[nt][r] *= alpha[r];

        #pragma unroll
        for (int nt = 0; nt < 4; ++nt)
            #pragma unroll
            for (int r = 0; r < 8; ++r)
                Ps[wave][lh * 8 + r][nt * 16 + l15] = (_Float16)s_acc[nt][r];
        __syncthreads();

        #pragma unroll
        for (int s = 0; s < 2; ++s) {
            F16x16 ap;
            ap.h[0] = *(const v8h*)&Ps[wave][l15][s * 32 + lh * 8];
            ap.h[1] = *(const v8h*)&Ps[wave][l15][s * 32 + 16 + lh * 8];
            #pragma unroll
            for (int nt = 0; nt < 4; ++nt) {
                F16x16 bv;
                bv.h[0] = *(const v8h*)&Vt[nt * 16 + l15][s * 32 + lh * 8];
                bv.h[1] = *(const v8h*)&Vt[nt * 16 + l15][s * 32 + 16 + lh * 8];
                o_acc[nt] = wmma_f16(ap.v, bv.v, o_acc[nt]);
            }
        }
    }

    #pragma unroll
    for (int nt = 0; nt < 4; ++nt) {
        #pragma unroll
        for (int r = 0; r < 8; ++r) {
            const int rl = lh * 8 + r;
            const float val = o_acc[nt][r] / l_i[r];
            const int trow = qt * 64 + wave * 16 + rl;
            o[(((size_t)b * SEQ_T + trow) * HEADS_H + h) * DH + nt * 16 + l15] =
                (_Float16)val;
        }
    }
}

// ---------------------------------------------------------------------------
// Host-side orchestration
// ---------------------------------------------------------------------------
extern "C" void kernel_launch(void* const* d_in, const int* in_sizes, int n_in,
                              void* d_out, int out_size, void* d_ws, size_t ws_size,
                              hipStream_t stream) {
    (void)in_sizes; (void)n_in; (void)out_size; (void)ws_size;

    const int*   ids   = (const int*)  d_in[0];
    const float* wte   = (const float*)d_in[1];
    const float* wpe   = (const float*)d_in[2];
    const float* ln1_g = (const float*)d_in[3];
    const float* ln1_b = (const float*)d_in[4];
    const float* wq    = (const float*)d_in[5];
    const float* bq    = (const float*)d_in[6];
    const float* wk    = (const float*)d_in[7];
    const float* bk    = (const float*)d_in[8];
    const float* wv    = (const float*)d_in[9];
    const float* bv    = (const float*)d_in[10];
    const float* wo    = (const float*)d_in[11];
    const float* bo    = (const float*)d_in[12];
    const float* ln2_g = (const float*)d_in[13];
    const float* ln2_b = (const float*)d_in[14];
    const float* w1    = (const float*)d_in[15];
    const float* b1    = (const float*)d_in[16];
    const float* w2    = (const float*)d_in[17];
    const float* b2    = (const float*)d_in[18];
    const float* lnf_g = (const float*)d_in[19];
    const float* lnf_b = (const float*)d_in[20];

    char* p = (char*)d_ws;
    auto alloc = [&](size_t bytes) -> void* {
        void* r = (void*)p;
        p += (bytes + 255) & ~(size_t)255;
        return r;
    };
    float*    xf   = (float*)   alloc((size_t)BT * DIM_D * 4);
    _Float16* h16  = (_Float16*)alloc((size_t)BT * DIM_D * 2);
    _Float16* q16  = (_Float16*)alloc((size_t)BT * DIM_D * 2);
    _Float16* k16  = (_Float16*)alloc((size_t)BT * DIM_D * 2);
    _Float16* v16  = (_Float16*)alloc((size_t)BT * DIM_D * 2);
    _Float16* o16  = (_Float16*)alloc((size_t)BT * DIM_D * 2);
    _Float16* ff16 = (_Float16*)alloc((size_t)BT * FFN_F * 2);
    _Float16* wq16 = (_Float16*)alloc((size_t)DIM_D * DIM_D * 2);  // [N][K]
    _Float16* wk16 = (_Float16*)alloc((size_t)DIM_D * DIM_D * 2);
    _Float16* wv16 = (_Float16*)alloc((size_t)DIM_D * DIM_D * 2);
    _Float16* wo16 = (_Float16*)alloc((size_t)DIM_D * DIM_D * 2);
    _Float16* w116 = (_Float16*)alloc((size_t)DIM_D * FFN_F * 2); // [F][D]
    _Float16* w216 = (_Float16*)alloc((size_t)FFN_F * DIM_D * 2); // [D][F]

    const size_t nDD = (size_t)DIM_D * DIM_D;
    const size_t nDF = (size_t)DIM_D * FFN_F;

    embed_kernel<<<BT, 256, 0, stream>>>(ids, wte, wpe, xf);

    for (int l = 0; l < L_LAYERS; ++l) {
        // W[K][N] fp32 -> Wt[N][K] f16
        cvt_t_kernel<<<dim3(DIM_D / 64, DIM_D / 64), 256, 0, stream>>>(
            wq + l * nDD, wq16, DIM_D, DIM_D);
        cvt_t_kernel<<<dim3(DIM_D / 64, DIM_D / 64), 256, 0, stream>>>(
            wk + l * nDD, wk16, DIM_D, DIM_D);
        cvt_t_kernel<<<dim3(DIM_D / 64, DIM_D / 64), 256, 0, stream>>>(
            wv + l * nDD, wv16, DIM_D, DIM_D);
        cvt_t_kernel<<<dim3(DIM_D / 64, DIM_D / 64), 256, 0, stream>>>(
            wo + l * nDD, wo16, DIM_D, DIM_D);
        cvt_t_kernel<<<dim3(FFN_F / 64, DIM_D / 64), 256, 0, stream>>>(
            w1 + l * nDF, w116, DIM_D, FFN_F);
        cvt_t_kernel<<<dim3(DIM_D / 64, FFN_F / 64), 256, 0, stream>>>(
            w2 + l * nDF, w216, FFN_F, DIM_D);

        ln_kernel<<<BT, 256, 0, stream>>>(xf, ln1_g + l * DIM_D, ln1_b + l * DIM_D,
                                          h16, nullptr);
        gemm_kernel<0, false, false, true>
            <<<dim3(DIM_D / 128, BT / 128), 256, 0, stream>>>(
            h16, wq16, bq + l * DIM_D, nullptr, nullptr, q16, BT, DIM_D, DIM_D);
        gemm_kernel<0, false, false, true>
            <<<dim3(DIM_D / 128, BT / 128), 256, 0, stream>>>(
            h16, wk16, bk + l * DIM_D, nullptr, nullptr, k16, BT, DIM_D, DIM_D);
        gemm_kernel<0, false, false, true>
            <<<dim3(DIM_D / 128, BT / 128), 256, 0, stream>>>(
            h16, wv16, bv + l * DIM_D, nullptr, nullptr, v16, BT, DIM_D, DIM_D);

        attn_kernel<<<dim3(SEQ_T / 64, HEADS_H, BATCH_B), 128, 0, stream>>>(
            q16, k16, v16, o16);

        gemm_kernel<0, true, true, false>
            <<<dim3(DIM_D / 128, BT / 128), 256, 0, stream>>>(
            o16, wo16, bo + l * DIM_D, xf, xf, nullptr, BT, DIM_D, DIM_D);

        ln_kernel<<<BT, 256, 0, stream>>>(xf, ln2_g + l * DIM_D, ln2_b + l * DIM_D,
                                          h16, nullptr);
        gemm_kernel<1, false, false, true>
            <<<dim3(FFN_F / 128, BT / 128), 256, 0, stream>>>(
            h16, w116, b1 + l * FFN_F, nullptr, nullptr, ff16, BT, FFN_F, DIM_D);
        gemm_kernel<0, true, true, false>
            <<<dim3(DIM_D / 128, BT / 128), 256, 0, stream>>>(
            ff16, w216, b2 + l * DIM_D, xf, xf, nullptr, BT, DIM_D, FFN_F);
    }

    ln_kernel<<<BT, 256, 0, stream>>>(xf, lnf_g, lnf_b, nullptr, (float*)d_out);
}
